// MultiheadAttention_41463614276147
// MI455X (gfx1250) — compile-verified, matrix operand order checked by reading, not run on hardware
//
#include <hip/hip_runtime.h>

// MI455X / gfx1250 — wave32, WMMA 16x16x32 f16 path.
// Pipeline: (1) QKV projection GEMMs (f32 -> f16, Q pre-scaled by 1/8, V stored
// transposed), (2) fused attention: scores WMMA -> softmax + top-k bisection +
// random-column mask -> a2@V WMMA with k-split + LDS reduction, (3) output
// projection GEMM (f32 out). All global fragment loads are b128.

#define SEQ    1024
#define DK     64
#define NHB    64      // HEADS*BATCH
#define TOPKN  51
#define MODSCL 1.2f

typedef _Float16 v16h __attribute__((ext_vector_type(16)));
typedef float    v8f  __attribute__((ext_vector_type(8)));

union Frag { v16h h; unsigned u[8]; };

__device__ __forceinline__ v8f vzero() {
  v8f v;
#pragma unroll
  for (int i = 0; i < 8; i++) v[i] = 0.0f;
  return v;
}

// 16x32 f16 fragment from LDS (dword loads; strides chosen for bank spread).
// ISA 7.12.2: lane l16 holds row/col l16; k-offsets = hf*8 + {0..7} and
// hf*8 + 16 + {0..7}, two f16 per VGPR (even k in [15:0]).
__device__ __forceinline__ Frag load_frag16(const _Float16* p, int ld, int l16, int hf) {
  Frag f;
  const _Float16* r = p + l16 * ld + hf * 8;
#pragma unroll
  for (int i = 0; i < 4; i++) f.u[i]     = *(const unsigned*)(r + 2 * i);
#pragma unroll
  for (int i = 0; i < 4; i++) f.u[4 + i] = *(const unsigned*)(r + 16 + 2 * i);
  return f;
}

// Same fragment from global memory with 16B-aligned rows -> two b128 loads.
__device__ __forceinline__ Frag load_frag16_g(const _Float16* p, int ld, int l16, int hf) {
  Frag f;
  const _Float16* r = p + l16 * ld + hf * 8;
  uint4 lo = *(const uint4*)(r);
  uint4 hi = *(const uint4*)(r + 16);
  f.u[0] = lo.x; f.u[1] = lo.y; f.u[2] = lo.z; f.u[3] = lo.w;
  f.u[4] = hi.x; f.u[5] = hi.y; f.u[6] = hi.z; f.u[7] = hi.w;
  return f;
}

__device__ __forceinline__ v8f wmma16(const Frag& a, const Frag& b, v8f c) {
  return __builtin_amdgcn_wmma_f32_16x16x32_f16(false, a.h, false, b.h,
                                                (short)0, c, false, false);
}

__device__ __forceinline__ float wred_max(float x) {
#pragma unroll
  for (int m = 16; m > 0; m >>= 1) x = fmaxf(x, __shfl_xor(x, m, 32));
  return x;
}
__device__ __forceinline__ float wred_sum(float x) {
#pragma unroll
  for (int m = 16; m > 0; m >>= 1) x += __shfl_xor(x, m, 32);
  return x;
}
__device__ __forceinline__ int wred_isum(int x) {
#pragma unroll
  for (int m = 16; m > 0; m >>= 1) x += __shfl_xor(x, m, 32);
  return x;
}

// ---------------- Kernel 1: QKV projections --------------------------------
// grid (128 Mtiles, 8 Ntiles, 3 projections), block 128 (4 waves).
__global__ __launch_bounds__(128) void qkv_kernel(
    const float* __restrict__ Xq, const float* __restrict__ Xkv,
    const float* __restrict__ Wq, const float* __restrict__ bq,
    const float* __restrict__ Wk, const float* __restrict__ bk,
    const float* __restrict__ Wv, const float* __restrict__ bv,
    _Float16* __restrict__ Qh, _Float16* __restrict__ Kh,
    _Float16* __restrict__ VhT) {
  __shared__ __align__(16) _Float16 sA[64 * 36];  // X tile  [m][k], padded
  __shared__ __align__(16) _Float16 sB[64 * 36];  // W tile  [n][k], padded
  const int t = threadIdx.x;
  const int w = t >> 5, lane = t & 31, l16 = lane & 15, hf = lane >> 4;
  const int proj = blockIdx.z;
  const int m0 = blockIdx.x * 64;
  const int n0 = blockIdx.y * 64;
  const float* X    = (proj == 0) ? Xq : Xkv;
  const float* W    = (proj == 0) ? Wq : ((proj == 1) ? Wk : Wv);
  const float* bias = (proj == 0) ? bq : ((proj == 1) ? bk : bv);

  v8f acc[4];
#pragma unroll
  for (int i = 0; i < 4; i++) acc[i] = vzero();

  for (int k0 = 0; k0 < 512; k0 += 32) {
#pragma unroll
    for (int j = 0; j < 4; j++) {       // 64x32 X tile: float4 loads
      int idx = t + 128 * j;            // 512 quads
      int r = idx >> 3, c4 = (idx & 7) * 4;
      float4 xv = *(const float4*)&X[(m0 + r) * 512 + k0 + c4];
      union { _Float16 h[4]; uint2 u; } pk;
      pk.h[0] = (_Float16)xv.x; pk.h[1] = (_Float16)xv.y;
      pk.h[2] = (_Float16)xv.z; pk.h[3] = (_Float16)xv.w;
      *(uint2*)&sA[r * 36 + c4] = pk.u;
    }
#pragma unroll
    for (int j = 0; j < 4; j++) {       // 32x64 W tile -> [n][k]
      int idx = t + 128 * j;
      int k = idx >> 4, n4 = (idx & 15) * 4;
      float4 wv = *(const float4*)&W[(k0 + k) * 512 + n0 + n4];
      sB[(n4 + 0) * 36 + k] = (_Float16)wv.x;
      sB[(n4 + 1) * 36 + k] = (_Float16)wv.y;
      sB[(n4 + 2) * 36 + k] = (_Float16)wv.z;
      sB[(n4 + 3) * 36 + k] = (_Float16)wv.w;
    }
    __syncthreads();
    Frag a = load_frag16(sA + (w * 16) * 36, 36, l16, hf);
#pragma unroll
    for (int nt = 0; nt < 4; nt++) {
      Frag b = load_frag16(sB + (nt * 16) * 36, 36, l16, hf);
      acc[nt] = wmma16(a, b, acc[nt]);
    }
    __syncthreads();
  }

  const float fs = (proj == 0) ? 0.125f : 1.0f;  // fold 1/sqrt(dk) into Q
#pragma unroll
  for (int nt = 0; nt < 4; nt++) {
    int c = n0 + nt * 16 + l16;
    float bval = bias[c];
    int h = c >> 6, d = c & 63;
#pragma unroll
    for (int i = 0; i < 8; i++) {
      int m = m0 + w * 16 + hf * 8 + i;       // C layout: row = hf*8+i
      int b = m >> 10, s = m & 1023;
      int n = h * 8 + b;                      // head-split index
      _Float16 hv = (_Float16)((acc[nt][i] + bval) * fs);
      if (proj == 2)      VhT[((n * 64 + d) << 10) + s] = hv;  // V transposed
      else if (proj == 1) Kh[((n << 10) + s) * 64 + d] = hv;
      else                Qh[((n << 10) + s) * 64 + d] = hv;
    }
  }
}

// ---------------- Kernel 2: fused attention --------------------------------
// grid (32 q-tiles, 64 head*batch), block 256 (8 waves), 32 query rows / WG.
__global__ __launch_bounds__(256) void attn_kernel(
    const _Float16* __restrict__ Qh, const _Float16* __restrict__ Kh,
    const _Float16* __restrict__ VhT, _Float16* __restrict__ Oh,
    const int* __restrict__ randidx, int nRand) {
  __shared__ __align__(16) float sScores[32 * 1025];  // padded; reused as f16 a2
  __shared__ __align__(16) float sPart[4 * 512];      // phase-4 partial sums
  __shared__ __align__(16) _Float16 sQ[32 * 66];
  __shared__ unsigned sMask[32];                      // 1024-bit random-col mask
  __shared__ float sRowScale[32];
  const int t = threadIdx.x;
  const int w = t >> 5, lane = t & 31, l16 = lane & 15, hf = lane >> 4;
  const int n = blockIdx.y;
  const int q0 = blockIdx.x * 32;

  // Phase 1: stage Q tile, build random-column bitmask.
  if (t < 32) sMask[t] = 0u;
#pragma unroll
  for (int j = 0; j < 8; j++) {
    int e = t + 256 * j;  // 32x64
    int r = e >> 6, d = e & 63;
    sQ[r * 66 + d] = Qh[((n << 10) + q0 + r) * 64 + d];
  }
  __syncthreads();
  if (t < nRand) {
    int idx = randidx[t] & 1023;
    atomicOr(&sMask[idx >> 5], 1u << (idx & 31));
  }

  // Phase 2: scores[32][1024] = Qtile @ K^T (Q already holds 1/8 scale).
  // Each wave: both 16-row strips x 8 column tiles; K fragment reused 2x.
  {
    Frag a00 = load_frag16(sQ,                66, l16, hf);  // rows 0-15,  k 0-31
    Frag a01 = load_frag16(sQ + 32,           66, l16, hf);  // rows 0-15,  k 32-63
    Frag a10 = load_frag16(sQ + 16 * 66,      66, l16, hf);  // rows 16-31, k 0-31
    Frag a11 = load_frag16(sQ + 16 * 66 + 32, 66, l16, hf);
    const _Float16* kbase = Kh + (n << 10) * 64;
#pragma unroll 1
    for (int i = 0; i < 8; i++) {
      int ct = w * 8 + i;
      const _Float16* kp = kbase + (ct * 16) * 64;  // B[k][s] = Kh[s][k]
      Frag b0 = load_frag16_g(kp, 64, l16, hf);
      Frag b1 = load_frag16_g(kp + 32, 64, l16, hf);
      v8f c0 = vzero(), c1 = vzero();
      c0 = wmma16(a00, b0, c0);
      c0 = wmma16(a01, b1, c0);
      c1 = wmma16(a10, b0, c1);
      c1 = wmma16(a11, b1, c1);
#pragma unroll
      for (int ii = 0; ii < 8; ii++) {
        sScores[(hf * 8 + ii) * 1025 + ct * 16 + l16]        = c0[ii];
        sScores[(16 + hf * 8 + ii) * 1025 + ct * 16 + l16]   = c1[ii];
      }
    }
  }
  __syncthreads();

  // Phase 3: per row: exp, top-k threshold by bisection on count, scale+mask,
  // write unnormalized a2 (f16, in-place), save 1/rowsum.
#pragma unroll 1
  for (int rr = 0; rr < 4; rr++) {
    int r = w * 4 + rr;
    float v[32];
#pragma unroll
    for (int j = 0; j < 32; j++) v[j] = sScores[r * 1025 + lane + 32 * j];
    float m = -3.4e38f;
#pragma unroll
    for (int j = 0; j < 32; j++) m = fmaxf(m, v[j]);
    m = wred_max(m);
#pragma unroll
    for (int j = 0; j < 32; j++) v[j] = __expf(v[j] - m);  // max elem -> 1.0
    float lo = 0.0f, hi = 1.0f;
#pragma unroll 1
    for (int it = 0; it < 25; it++) {  // find 51st-largest value
      float mid = 0.5f * (lo + hi);
      int cnt = 0;
#pragma unroll
      for (int j = 0; j < 32; j++) cnt += (v[j] >= mid) ? 1 : 0;
      cnt = wred_isum(cnt);
      if (cnt >= TOPKN) lo = mid; else hi = mid;
    }
    float thr = lo;
    float sum = 0.0f;
    _Float16* a2 = (_Float16*)sScores + r * 2050;  // in-place, stride 4100 B
#pragma unroll
    for (int j = 0; j < 32; j++) {
      unsigned mb = sMask[j];  // col = lane + 32*j -> bit `lane` of word j
      bool sel = (v[j] >= thr) || ((mb >> lane) & 1u);
      float av = sel ? v[j] * MODSCL : v[j];
      sum += av;
      a2[lane + 32 * j] = (_Float16)av;
    }
    sum = wred_sum(sum);
    if (lane == 0) sRowScale[r] = 1.0f / sum;
  }
  __syncthreads();

  // Phase 4: out(32x64) = a2(32x1024) @ V(1024x64). Wave pair (2g, 2g+1)
  // shares column tile no = g*16 and splits the k-range; each V fragment is
  // reused for both 16-row output tiles. Partials reduced through LDS.
  {
    const int no = (w >> 1) * 16, kh = w & 1;
    const _Float16* a2b0 = (const _Float16*)sScores;             // rows 0-15
    const _Float16* a2b1 = (const _Float16*)sScores + 16 * 2050; // rows 16-31
    const _Float16* vbase = VhT + ((n * 64 + no) << 10);         // [d][s]
    v8f c0 = vzero(), c1 = vzero();
#pragma unroll 2
    for (int ks = kh * 512; ks < kh * 512 + 512; ks += 32) {
      Frag b  = load_frag16_g(vbase + ks, 1024, l16, hf);
      Frag a0 = load_frag16(a2b0 + ks, 2050, l16, hf);
      c0 = wmma16(a0, b, c0);
      Frag a1 = load_frag16(a2b1 + ks, 2050, l16, hf);
      c1 = wmma16(a1, b, c1);
    }
    if (kh) {
      float* pp = sPart + (w >> 1) * 512;
#pragma unroll
      for (int i = 0; i < 8; i++) {
        pp[i * 32 + lane]       = c0[i];
        pp[256 + i * 32 + lane] = c1[i];
      }
    }
    __syncthreads();
    if (!kh) {
      const float* pp = sPart + (w >> 1) * 512;
#pragma unroll
      for (int i = 0; i < 8; i++) {
        c0[i] += pp[i * 32 + lane];
        c1[i] += pp[256 + i * 32 + lane];
      }
#pragma unroll
      for (int i = 0; i < 8; i++) {
        int r0 = hf * 8 + i;
        Oh[((n << 10) + q0 + r0) * 64 + no + l16] =
            (_Float16)(c0[i] * sRowScale[r0]);
        int r1 = 16 + hf * 8 + i;
        Oh[((n << 10) + q0 + r1) * 64 + no + l16] =
            (_Float16)(c1[i] * sRowScale[r1]);
      }
    }
  }
}

// ---------------- Kernel 3: output projection ------------------------------
__global__ __launch_bounds__(128) void oproj_kernel(
    const _Float16* __restrict__ Oh, const float* __restrict__ Wo,
    const float* __restrict__ bo, float* __restrict__ out) {
  __shared__ __align__(16) _Float16 sA[64 * 36];
  __shared__ __align__(16) _Float16 sB[64 * 36];
  const int t = threadIdx.x;
  const int w = t >> 5, lane = t & 31, l16 = lane & 15, hf = lane >> 4;
  const int m0 = blockIdx.x * 64;
  const int n0 = blockIdx.y * 64;
  v8f acc[4];
#pragma unroll
  for (int i = 0; i < 4; i++) acc[i] = vzero();

  for (int k0 = 0; k0 < 512; k0 += 32) {
#pragma unroll
    for (int j = 0; j < 8; j++) {   // gather merged-heads rows from Oh, u32 pairs
      int e = t + 128 * j;          // 1024 f16-pairs
      int r = e >> 4, c2 = (e & 15) * 2;
      int m = m0 + r, b = m >> 10, s = m & 1023;
      int k = k0 + c2, h = k >> 6, d = k & 63;
      *(unsigned*)&sA[r * 36 + c2] =
          *(const unsigned*)&Oh[(((h * 8 + b) << 10) + s) * 64 + d];
    }
#pragma unroll
    for (int j = 0; j < 4; j++) {
      int idx = t + 128 * j;
      int k = idx >> 4, n4 = (idx & 15) * 4;
      float4 wv = *(const float4*)&Wo[(k0 + k) * 512 + n0 + n4];
      sB[(n4 + 0) * 36 + k] = (_Float16)wv.x;
      sB[(n4 + 1) * 36 + k] = (_Float16)wv.y;
      sB[(n4 + 2) * 36 + k] = (_Float16)wv.z;
      sB[(n4 + 3) * 36 + k] = (_Float16)wv.w;
    }
    __syncthreads();
    Frag a = load_frag16(sA + (w * 16) * 36, 36, l16, hf);
#pragma unroll
    for (int nt = 0; nt < 4; nt++) {
      Frag b = load_frag16(sB + (nt * 16) * 36, 36, l16, hf);
      acc[nt] = wmma16(a, b, acc[nt]);
    }
    __syncthreads();
  }
#pragma unroll
  for (int nt = 0; nt < 4; nt++) {
    int c = n0 + nt * 16 + l16;
    float bval = bo[c];
#pragma unroll
    for (int i = 0; i < 8; i++) {
      int m = m0 + w * 16 + hf * 8 + i;
      out[m * 512 + c] = acc[nt][i] + bval;
    }
  }
}

extern "C" void kernel_launch(void* const* d_in, const int* in_sizes, int n_in,
                              void* d_out, int out_size, void* d_ws, size_t ws_size,
                              hipStream_t stream) {
  const float* Xq  = (const float*)d_in[0];
  const float* Xkv = (const float*)d_in[1];
  const float* Wq  = (const float*)d_in[2];
  const float* bq  = (const float*)d_in[3];
  const float* Wk  = (const float*)d_in[4];
  const float* bk  = (const float*)d_in[5];
  const float* Wv  = (const float*)d_in[6];
  const float* bv  = (const float*)d_in[7];
  const float* Wo  = (const float*)d_in[8];
  const float* bo  = (const float*)d_in[9];
  const int* randidx = (const int*)d_in[10];
  const int nRand = in_sizes[10];
  float* out = (float*)d_out;

  // Workspace: Qh | Kh | VhT | Oh, each 64*1024*64 f16 = 8 MB (32 MB total).
  _Float16* Qh  = (_Float16*)d_ws;
  _Float16* Kh  = Qh + (size_t)NHB * SEQ * DK;
  _Float16* VhT = Kh + (size_t)NHB * SEQ * DK;
  _Float16* Oh  = VhT + (size_t)NHB * SEQ * DK;

  qkv_kernel<<<dim3(128, 8, 3), 128, 0, stream>>>(Xq, Xkv, Wq, bq, Wk, bk, Wv, bv,
                                                  Qh, Kh, VhT);
  attn_kernel<<<dim3(32, 64), 256, 0, stream>>>(Qh, Kh, VhT, Oh, randidx, nRand);
  oproj_kernel<<<dim3(128, 8), 128, 0, stream>>>(Oh, Wo, bo, out);
}
